// BertLinearSelfAttention_50775103373468
// MI455X (gfx1250) — matrix operand order repaired and checked by reading, not compiled, not run on hardware
//
#include <hip/hip_runtime.h>
#include <stdint.h>

// ---------------------------------------------------------------------------
// Linear attention, B=4 S=4096 D=1024 H=16 HD=64.  f16 WMMA, f32 accum.
//   P0a: hs f32 -> f16 (row-major)
//   P0b: W f32 -> W^T f16
//   P1 : qkv GEMM (async-to-LDS double buffered); q -> [BS][D],
//        k/v -> transposed [bh][d][S] f16
//   P2 : k_sum[bh][d]
//   P3 : kv^T[bh][e][d]  -- tiles staged by the Tensor Data Mover
//   P4 : out = (q_feat @ kv) / max(q_feat . k_sum, eps)  -- TDM staging
// ---------------------------------------------------------------------------

typedef __attribute__((ext_vector_type(16))) _Float16 v16h;
typedef __attribute__((ext_vector_type(8)))  _Float16 v8h;
typedef __attribute__((ext_vector_type(8)))  float    v8f;
typedef __attribute__((ext_vector_type(4)))  uint32_t u32x4;
typedef __attribute__((ext_vector_type(8)))  uint32_t u32x8;

#define Bz  4
#define Sz  4096
#define Dz  1024
#define Hz  16
#define HDz 64
#define BSz (Bz * Sz)          // 16384
#define EPSv 1e-6f

__device__ __forceinline__ v8f wmma32(v16h a, v16h b, v8f c) {
  return __builtin_amdgcn_wmma_f32_16x16x32_f16(
      false, a, false, b, (short)0, c, false, false);
}

__device__ __forceinline__ v16h frag2(const _Float16* p0, const _Float16* p1) {
  v8h lo = *(const v8h*)p0;
  v8h hi = *(const v8h*)p1;
  v16h r;
#pragma unroll
  for (int i = 0; i < 8; ++i) { r[i] = lo[i]; r[i + 8] = hi[i]; }
  return r;
}

// --- async per-lane DMA (used in the big QKV GEMM) -------------------------
__device__ __forceinline__ void async_copy16(const _Float16* lds_dst,
                                             const _Float16* gsrc) {
  asm volatile("global_load_async_to_lds_b128 %0, %1, off"
               :: "v"((uint32_t)(uintptr_t)lds_dst),
                  "v"((uint64_t)(uintptr_t)gsrc)
               : "memory");
}
__device__ __forceinline__ void wait_async0() {
  asm volatile("s_wait_asynccnt 0x0" ::: "memory");
}
__device__ __forceinline__ void wait_async3() {
  asm volatile("s_wait_asynccnt 0x3" ::: "memory");
}

// --- Tensor Data Mover: 2D f16 tile, global -> LDS -------------------------
// Rows of tile_x f16 elements (tile_x*2 bytes); when pad!=0, 16B of LDS
// padding is inserted after every 128B stored (pad_interval=4 -> 32 DWORDs,
// pad_amount=3 -> 4 DWORDs), matching the [rows][72]-half LDS layout.
__device__ __forceinline__ void tdm_load_2d(uint32_t lds_addr, const void* g,
                                            uint32_t tile_x, uint32_t tile_rows,
                                            uint32_t stride_elems,
                                            uint32_t tensor_x, uint32_t tensor_y,
                                            int pad) {
  uint64_t ga = (uint64_t)(uintptr_t)g;
  u32x4 g0;
  g0[0] = 1u;                                   // count=1, user descriptor
  g0[1] = lds_addr;                             // lds_addr[31:0]
  g0[2] = (uint32_t)ga;                         // global_addr[31:0]
  g0[3] = ((uint32_t)(ga >> 32) & 0x01FFFFFFu)  // global_addr[56:32]
          | (2u << 30);                         // type = 2 ("image")
  u32x8 g1;
  g1[0] = (1u << 16)                                    // data_size = 2B
        | (pad ? ((1u << 20) | (4u << 22) | (3u << 25)) : 0u);
  g1[1] = (tensor_x & 0xFFFFu) << 16;                   // tensor_dim0[15:0]
  g1[2] = (tensor_x >> 16) | ((tensor_y & 0xFFFFu) << 16);
  g1[3] = (tensor_y >> 16) | (tile_x << 16);            // tile_dim0
  g1[4] = tile_rows;                                    // tile_dim1 (tile_dim2=0)
  g1[5] = stride_elems;                                 // tensor_dim0_stride lo
  g1[6] = 0u;
  g1[7] = 0u;
  asm volatile("tensor_load_to_lds %0, %1" :: "s"(g0), "s"(g1) : "memory");
}

// ---------------------------------------------------------------------------
// P0a: straight f32 -> f16, 8 elem/thread
// ---------------------------------------------------------------------------
__global__ void cvt_f32_to_f16(const float* __restrict__ src,
                               _Float16* __restrict__ dst, int n8) {
  int i = blockIdx.x * blockDim.x + threadIdx.x;
  if (i >= n8) return;
  float4 a = ((const float4*)src)[2 * i];
  float4 b = ((const float4*)src)[2 * i + 1];
  v8h o;
  o[0] = (_Float16)a.x; o[1] = (_Float16)a.y;
  o[2] = (_Float16)a.z; o[3] = (_Float16)a.w;
  o[4] = (_Float16)b.x; o[5] = (_Float16)b.y;
  o[6] = (_Float16)b.z; o[7] = (_Float16)b.w;
  ((v8h*)dst)[i] = o;
}

// ---------------------------------------------------------------------------
// P0b: W[K][N] f32 -> WT[N][K] f16, 16x16 LDS tile transpose
// ---------------------------------------------------------------------------
__global__ void cvt_transpose_f16(const float* __restrict__ W,
                                  _Float16* __restrict__ WT) {
  __shared__ float t[16][17];
  const int n0 = blockIdx.x * 16, k0 = blockIdx.y * 16;
  const int tx = threadIdx.x, ty = threadIdx.y;
  t[ty][tx] = W[(size_t)(k0 + ty) * Dz + n0 + tx];
  __syncthreads();
  WT[(size_t)(n0 + ty) * Dz + k0 + tx] = (_Float16)t[tx][ty];
}

// ---------------------------------------------------------------------------
// P1: act(A[M,K] * W[K,N] + bias).  Async-to-LDS, double buffered.
// MODE: 1 = q (elu+1, plain), 2 = k (elu+1, transposed), 0 = v (transposed).
// ---------------------------------------------------------------------------
template <int MODE>
__global__ void qkv_gemm(const _Float16* __restrict__ A,
                         const _Float16* __restrict__ WT,
                         const float* __restrict__ bias,
                         _Float16* __restrict__ out) {
  __shared__ alignas(16) _Float16 Al[2][128][40];
  __shared__ alignas(16) _Float16 Bt[2][64][40];

  const int m0   = blockIdx.x * 128;
  const int n0   = blockIdx.y * 64;
  const int tid  = threadIdx.x;
  const int wave = tid >> 5, lane = tid & 31;
  const int wm = wave & 3, wn = wave >> 2;
  const int g = lane >> 4, MN = lane & 15;

  auto issue = [&](int buf, int k0) {
#pragma unroll
    for (int i = 0; i < 2; ++i) {
      int idx = tid + i * 256;
      int r = idx >> 2, seg = idx & 3;
      async_copy16(&Al[buf][r][seg * 8],
                   A + (size_t)(m0 + r) * Dz + k0 + seg * 8);
    }
    {
      int r = tid >> 2, seg = tid & 3;
      async_copy16(&Bt[buf][r][seg * 8],
                   WT + (size_t)(n0 + r) * Dz + k0 + seg * 8);
    }
  };

  v8f acc[2][2] = {};
  issue(0, 0);

  for (int kt = 0; kt < Dz / 32; ++kt) {
    const int buf = kt & 1;
    if (kt + 1 < Dz / 32) { issue(buf ^ 1, (kt + 1) * 32); wait_async3(); }
    else                  { wait_async0(); }
    __syncthreads();

    v16h af[2], bf[2];
#pragma unroll
    for (int ti = 0; ti < 2; ++ti) {
      const _Float16* ar = &Al[buf][wm * 32 + ti * 16 + MN][0];
      af[ti] = frag2(ar + g * 8, ar + 16 + g * 8);
    }
#pragma unroll
    for (int tj = 0; tj < 2; ++tj) {
      const _Float16* br = &Bt[buf][wn * 32 + tj * 16 + MN][0];
      bf[tj] = frag2(br + g * 8, br + 16 + g * 8);
    }
#pragma unroll
    for (int ti = 0; ti < 2; ++ti)
#pragma unroll
      for (int tj = 0; tj < 2; ++tj)
        acc[ti][tj] = wmma32(af[ti], bf[tj], acc[ti][tj]);
    __syncthreads();
  }

#pragma unroll
  for (int ti = 0; ti < 2; ++ti)
#pragma unroll
    for (int tj = 0; tj < 2; ++tj)
#pragma unroll
      for (int jj = 0; jj < 8; ++jj) {
        int row = m0 + wm * 32 + ti * 16 + jj + 8 * g;
        int col = n0 + wn * 32 + tj * 16 + MN;
        float v = acc[ti][tj][jj] + bias[col];
        if (MODE != 0) v = (v > 0.0f) ? (v + 1.0f) : __expf(v);
        _Float16 hv = (_Float16)v;
        if (MODE == 1) {
          out[(size_t)row * Dz + col] = hv;
        } else {
          int b = row >> 12, s = row & (Sz - 1);
          int h = col >> 6, d = col & (HDz - 1);
          out[(((size_t)(b * Hz + h) * HDz + d) << 12) + s] = hv;
        }
      }
}

// ---------------------------------------------------------------------------
// P2: k_sum[bh][d] = sum_s kfT[bh][d][s]
// ---------------------------------------------------------------------------
__global__ void ksum_kernel(const _Float16* __restrict__ kfT,
                            float* __restrict__ ksum) {
  __shared__ float red[256];
  const int bh = blockIdx.x, tid = threadIdx.x;
  const int d = tid >> 2, t4 = tid & 3;
  const _Float16* base = kfT + ((size_t)(bh * HDz + d) << 12);
  float acc = 0.0f;
  for (int c = t4; c < Sz / 8; c += 4) {
    v8h x = *(const v8h*)(base + c * 8);
#pragma unroll
    for (int j = 0; j < 8; ++j) acc += (float)x[j];
  }
  red[tid] = acc;
  __syncthreads();
  if (t4 == 0)
    ksum[bh * HDz + d] = red[tid] + red[tid + 1] + red[tid + 2] + red[tid + 3];
}

// ---------------------------------------------------------------------------
// P3: kvT[bh][e][d] = sum_s kf[s,d] * v[s,e].  One block per (b,h).
// Tiles staged by the Tensor Data Mover (wave 0 issues, TENSORcnt waits),
// double-buffered against the WMMA chain.
// ---------------------------------------------------------------------------
__global__ void kv_gemm(const _Float16* __restrict__ kfT,
                        const _Float16* __restrict__ vfT,
                        _Float16* __restrict__ kvT) {
  __shared__ alignas(16) _Float16 Kt[2][64][72];  // [d][s_local] (+TDM pad)
  __shared__ alignas(16) _Float16 Vt[2][64][72];  // [e][s_local]

  const int bh = blockIdx.x;
  const int tid = threadIdx.x;
  const int wave = tid >> 5, lane = tid & 31;
  const int g = lane >> 4, MN = lane & 15;
  const int dRow0 = (wave & 3) * 16;
  const int eCol0 = (wave >> 2) * 32;
  const _Float16* kbase = kfT + ((size_t)bh << 18);  // 64*4096
  const _Float16* vbase = vfT + ((size_t)bh << 18);

  auto issue = [&](int buf, int s0) {   // wave 0 only: 2 TDM ops
    tdm_load_2d((uint32_t)(uintptr_t)&Kt[buf][0][0], kbase + s0,
                /*tile_x=*/64, /*rows=*/64, /*stride=*/Sz,
                /*tensor_x=*/Sz, /*tensor_y=*/64, /*pad=*/1);
    tdm_load_2d((uint32_t)(uintptr_t)&Vt[buf][0][0], vbase + s0,
                64, 64, Sz, Sz, 64, 1);
  };

  v8f acc[2] = {};
  if (wave == 0) issue(0, 0);

  for (int c = 0; c < Sz / 64; ++c) {
    const int buf = c & 1;
    if (wave == 0) {
      if (c + 1 < Sz / 64) {
        issue(buf ^ 1, (c + 1) * 64);
        __builtin_amdgcn_s_wait_tensorcnt((short)2);  // current pair done
      } else {
        __builtin_amdgcn_s_wait_tensorcnt((short)0);
      }
    }
    __syncthreads();
#pragma unroll
    for (int k0 = 0; k0 < 64; k0 += 32) {
      const _Float16* ar = &Kt[buf][dRow0 + MN][0];
      v16h a = frag2(ar + k0 + g * 8, ar + k0 + 16 + g * 8);
#pragma unroll
      for (int tj = 0; tj < 2; ++tj) {
        const _Float16* br = &Vt[buf][eCol0 + tj * 16 + MN][0];
        v16h bfr = frag2(br + k0 + g * 8, br + k0 + 16 + g * 8);
        acc[tj] = wmma32(a, bfr, acc[tj]);
      }
    }
    __syncthreads();
  }

#pragma unroll
  for (int tj = 0; tj < 2; ++tj)
#pragma unroll
    for (int jj = 0; jj < 8; ++jj) {
      int d = dRow0 + jj + 8 * g;
      int e = eCol0 + tj * 16 + MN;
      kvT[(size_t)bh * HDz * HDz + e * HDz + d] = (_Float16)acc[tj][jj];
    }
}

// ---------------------------------------------------------------------------
// P4: out = (q_feat @ kv) / max(q_feat . k_sum, eps).  TDM one-shot staging.
// ---------------------------------------------------------------------------
__global__ void out_gemm(const _Float16* __restrict__ qf,
                         const _Float16* __restrict__ kvT,
                         const float* __restrict__ ksum,
                         float* __restrict__ out) {
  __shared__ alignas(16) _Float16 Qa[128][72];   // [s_local][d] (+TDM pad)
  __shared__ alignas(16) _Float16 KVt[64][72];   // [e][d]
  __shared__ float ksl[64];
  __shared__ float den[128];

  const int m0 = blockIdx.x * 128;
  const int bh = blockIdx.y, b = bh >> 4, h = bh & 15;
  const int tid = threadIdx.x;
  const int wave = tid >> 5, lane = tid & 31;
  const int wm = wave & 3, wn = wave >> 2;
  const int g = lane >> 4, MN = lane & 15;

  if (wave == 0) {
    tdm_load_2d((uint32_t)(uintptr_t)&Qa[0][0],
                qf + (size_t)(b * Sz + m0) * Dz + h * HDz,
                /*tile_x=*/64, /*rows=*/128, /*stride=*/Dz,
                /*tensor_x=*/Dz, /*tensor_y=*/128, /*pad=*/1);
    tdm_load_2d((uint32_t)(uintptr_t)&KVt[0][0],
                kvT + (size_t)bh * HDz * HDz,
                64, 64, HDz, HDz, 64, 1);
    __builtin_amdgcn_s_wait_tensorcnt((short)0);
  }
  if (tid < 64) ksl[tid] = ksum[bh * HDz + tid];
  __syncthreads();

  if (tid < 128) {
    float s = 0.0f;
#pragma unroll 8
    for (int d = 0; d < HDz; ++d) s += (float)Qa[tid][d] * ksl[d];
    den[tid] = fmaxf(s, EPSv);
  }
  __syncthreads();

  v8f acc[2][2] = {};
#pragma unroll
  for (int k0 = 0; k0 < 64; k0 += 32) {
    v16h af[2], bf[2];
#pragma unroll
    for (int ti = 0; ti < 2; ++ti) {
      const _Float16* ar = &Qa[wm * 32 + ti * 16 + MN][0];
      af[ti] = frag2(ar + k0 + g * 8, ar + k0 + 16 + g * 8);
    }
#pragma unroll
    for (int tj = 0; tj < 2; ++tj) {
      const _Float16* br = &KVt[wn * 32 + tj * 16 + MN][0];
      bf[tj] = frag2(br + k0 + g * 8, br + k0 + 16 + g * 8);
    }
#pragma unroll
    for (int ti = 0; ti < 2; ++ti)
#pragma unroll
      for (int tj = 0; tj < 2; ++tj)
        acc[ti][tj] = wmma32(af[ti], bf[tj], acc[ti][tj]);
  }

#pragma unroll
  for (int ti = 0; ti < 2; ++ti)
#pragma unroll
    for (int tj = 0; tj < 2; ++tj)
#pragma unroll
      for (int jj = 0; jj < 8; ++jj) {
        int rl = wm * 32 + ti * 16 + jj + 8 * g;
        int e = wn * 32 + tj * 16 + MN;
        out[(size_t)(b * Sz + m0 + rl) * Dz + h * HDz + e] =
            acc[ti][tj][jj] / den[rl];
      }
}

// ---------------------------------------------------------------------------
// launch
// ---------------------------------------------------------------------------
extern "C" void kernel_launch(void* const* d_in, const int* in_sizes, int n_in,
                              void* d_out, int out_size, void* d_ws, size_t ws_size,
                              hipStream_t stream) {
  const float* hs = (const float*)d_in[0];
  const float* Wq = (const float*)d_in[1];
  const float* bq = (const float*)d_in[2];
  const float* Wk = (const float*)d_in[3];
  const float* bk = (const float*)d_in[4];
  const float* Wv = (const float*)d_in[5];
  const float* bv = (const float*)d_in[6];
  float* out = (float*)d_out;
  (void)in_sizes; (void)n_in; (void)out_size; (void)ws_size;

  char* ws = (char*)d_ws;
  size_t off = 0;
  auto take = [&](size_t bytes) {
    void* p = ws + off;
    off = (off + bytes + 255) & ~(size_t)255;
    return p;
  };
  _Float16* hsF = (_Float16*)take((size_t)BSz * Dz * 2);
  _Float16* w0T = (_Float16*)take((size_t)Dz * Dz * 2);
  _Float16* w1T = (_Float16*)take((size_t)Dz * Dz * 2);
  _Float16* w2T = (_Float16*)take((size_t)Dz * Dz * 2);
  _Float16* qf  = (_Float16*)take((size_t)BSz * Dz * 2);   // [BS][D]
  _Float16* kfT = (_Float16*)take((size_t)BSz * Dz * 2);   // [bh][d][S]
  _Float16* vfT = (_Float16*)take((size_t)BSz * Dz * 2);   // [bh][e][S]
  _Float16* kvT = (_Float16*)take((size_t)Bz * Hz * HDz * HDz * 2);  // [bh][e][d]
  float*    ks  = (float*)take((size_t)Bz * Hz * HDz * 4);

  // P0
  {
    int n8 = BSz * Dz / 8;
    cvt_f32_to_f16<<<(n8 + 255) / 256, 256, 0, stream>>>(hs, hsF, n8);
    dim3 tg(64, 64), tb(16, 16);
    cvt_transpose_f16<<<tg, tb, 0, stream>>>(Wq, w0T);
    cvt_transpose_f16<<<tg, tb, 0, stream>>>(Wk, w1T);
    cvt_transpose_f16<<<tg, tb, 0, stream>>>(Wv, w2T);
  }

  // P1
  {
    dim3 grid(BSz / 128, Dz / 64);  // 128 x 16
    qkv_gemm<1><<<grid, 256, 0, stream>>>(hsF, w0T, bq, qf);   // q
    qkv_gemm<2><<<grid, 256, 0, stream>>>(hsF, w1T, bk, kfT);  // k
    qkv_gemm<0><<<grid, 256, 0, stream>>>(hsF, w2T, bv, vfT);  // v
  }

  // P2
  ksum_kernel<<<Bz * Hz, 256, 0, stream>>>(kfT, ks);

  // P3 (TDM staged)
  kv_gemm<<<Bz * Hz, 256, 0, stream>>>(kfT, vfT, kvT);

  // P4 (TDM staged)
  {
    dim3 grid(Sz / 128, Bz * Hz);   // 32 x 64
    out_gemm<<<grid, 256, 0, stream>>>(qf, kvT, ks, out);
  }
}
